// VectorQuantizer_15839839387913
// MI455X (gfx1250) — matrix operand order repaired and checked by reading, not compiled, not run on hardware
//
#include <hip/hip_runtime.h>
#include <hip/hip_bf16.h>

typedef __attribute__((ext_vector_type(16))) _Float16 v16h;
typedef __attribute__((ext_vector_type(8)))  _Float16 v8h;
typedef __attribute__((ext_vector_type(4)))  _Float16 v4h;
typedef __attribute__((ext_vector_type(8)))  float    v8f;

#define DIM 512
#define LDS_STRIDE 520   // 512 + 8 halves pad: breaks the all-lanes-same-bank pattern

// ---- CDNA5 async global->LDS staging (guarded; falls back to sync copy) ----
#if defined(__HIP_DEVICE_COMPILE__) && defined(__gfx1250__) && \
    __has_builtin(__builtin_amdgcn_global_load_async_to_lds_b128)
#define VQ_ASYNC_STAGE 1
// Exact param types per clang diagnostic: AS1/AS3 pointers to a 16-byte
// GCC-style int vector.
typedef int vq_b128 __attribute__((vector_size(16)));
typedef __attribute__((address_space(1))) vq_b128 vq_b128_g;
typedef __attribute__((address_space(3))) vq_b128 vq_b128_l;
// Integer round-trip avoids generic->AS cast legality questions:
// AS1 value == 64-bit VA; AS3 value == low 32 bits of the generic LDS address.
__device__ __forceinline__ vq_b128_g* vq_as1(const void* p) {
  return (vq_b128_g*)(size_t)p;
}
__device__ __forceinline__ vq_b128_l* vq_as3(const void* p) {
  return (vq_b128_l*)(unsigned int)(size_t)p;
}
#endif

__device__ __forceinline__ void vq_wait_async() {
#if defined(VQ_ASYNC_STAGE)
#if __has_builtin(__builtin_amdgcn_s_wait_asynccnt)
  __builtin_amdgcn_s_wait_asynccnt(0);
#else
  asm volatile("s_wait_asynccnt 0x0" ::: "memory");
#endif
#endif
}

// IGroupLP pipeline: keep LDS B-fragment reads 4 fragments ahead of the WMMA
// that consumes them (~3 WMMA latencies of cover > LDS latency).
// mask 0x008 = MFMA/WMMA, 0x100 = DS read.
__device__ __forceinline__ void vq_sched_pipeline() {
#if defined(__HIP_DEVICE_COMPILE__) && \
    __has_builtin(__builtin_amdgcn_sched_group_barrier)
  __builtin_amdgcn_sched_group_barrier(0x100, 8, 0);   // frags 0-3 (8 ds reads)
#pragma unroll
  for (int i = 0; i < 12; ++i) {
    __builtin_amdgcn_sched_group_barrier(0x008, 1, 0); // wmma(i)
    __builtin_amdgcn_sched_group_barrier(0x100, 2, 0); // frag i+4 loads
  }
  __builtin_amdgcn_sched_group_barrier(0x008, 4, 0);   // wmma(12..15)
#endif
}

// ---------------------------------------------------------------------------
// Kernel 1/2: row-wise L2 normalize. Writes f32 rows (optional) + f16 rows.
// One wave per row; lane handles 4 float4 chunks (coalesced).
// ---------------------------------------------------------------------------
__global__ void __launch_bounds__(256) vq_norm_rows(
    const float* __restrict__ in, float* __restrict__ out_f32,
    _Float16* __restrict__ out_f16, int nrows) {
  const int wave = threadIdx.x >> 5;
  const int lane = threadIdx.x & 31;
  const int row  = blockIdx.x * 8 + wave;
  if (row >= nrows) return;

  const float4* src = (const float4*)(in + (size_t)row * DIM);
  float4 r[4];
  float ss = 0.f;
#pragma unroll
  for (int i = 0; i < 4; ++i) {
    r[i] = src[lane + 32 * i];
    ss += r[i].x * r[i].x + r[i].y * r[i].y + r[i].z * r[i].z + r[i].w * r[i].w;
  }
#pragma unroll
  for (int m = 16; m >= 1; m >>= 1) ss += __shfl_xor(ss, m, 32);
  const float scale = 1.0f / fmaxf(sqrtf(ss), 1e-12f);

  _Float16* o16 = out_f16 + (size_t)row * DIM;
#pragma unroll
  for (int i = 0; i < 4; ++i) {
    float4 v = r[i];
    v.x *= scale; v.y *= scale; v.z *= scale; v.w *= scale;
    if (out_f32) ((float4*)(out_f32 + (size_t)row * DIM))[lane + 32 * i] = v;
    v4h h;
    h[0] = (_Float16)v.x; h[1] = (_Float16)v.y;
    h[2] = (_Float16)v.z; h[3] = (_Float16)v.w;
    *(v4h*)(o16 + 4 * (lane + 32 * i)) = h;
  }
}

// ---------------------------------------------------------------------------
// Kernel 3: sims = flat_f16 @ cb_f16^T with per-row argmax, via WMMA.
// Block = 256 threads = 8 waves; each wave owns 16 x-rows (WMMA M dim),
// iterates over M/16 codebook tiles (WMMA N dim), K = 512 = 16 x 32.
// A fragments register-resident for the whole M loop; B tile double-buffered
// in LDS via async DMA; B fragments pipelined in VGPRs via sched_group_barrier.
// ---------------------------------------------------------------------------
__global__ void __launch_bounds__(256) vq_gemm_argmax(
    const _Float16* __restrict__ A,   // [N, 512] normalized x, f16
    const _Float16* __restrict__ B,   // [M, 512] normalized codebook, f16
    int* __restrict__ best_idx, int M) {
  __shared__ _Float16 lds[2][16 * LDS_STRIDE];

  const int tid  = threadIdx.x;
  const int wave = tid >> 5;
  const int lane = tid & 31;
  const int n16  = lane & 15;
  const bool hi  = lane >= 16;
  const int row_base = blockIdx.x * 128 + wave * 16;

  // --- A fragments: ISA 16-bit A layout (16x32 per K-tile).
  // lanes 0-15: halves = K{0..7} then K{16..23}; lanes 16-31: +8.
  v16h a[16];
  {
    const _Float16* arow = A + (size_t)(row_base + n16) * DIM + (hi ? 8 : 0);
#pragma unroll
    for (int kt = 0; kt < 16; ++kt) {
      v8h lo8 = *(const v8h*)(arow + kt * 32);
      v8h hi8 = *(const v8h*)(arow + kt * 32 + 16);
      a[kt] = __builtin_shufflevector(lo8, hi8,
          0, 1, 2, 3, 4, 5, 6, 7, 8, 9, 10, 11, 12, 13, 14, 15);
    }
  }

  float best[8];
  int   bidx[8];
#pragma unroll
  for (int r = 0; r < 8; ++r) { best[r] = -3.402823e38f; bidx[r] = 0; }

  // Stage one 16-row codebook tile (16 KB) into LDS: 256 threads x 64 B.
  auto stage = [&](int tile, int buf) {
    const int rr = tid >> 4;          // 0..15: tile row
    const int cc = (tid & 15) * 32;   // column base (halves)
    const _Float16* g = B + (size_t)(tile * 16 + rr) * DIM + cc;
    _Float16* s = &lds[buf][rr * LDS_STRIDE + cc];
#if defined(VQ_ASYNC_STAGE)
    // ASYNCcnt-tracked DMA; same INST_OFFSET applies to both addresses.
    __builtin_amdgcn_global_load_async_to_lds_b128(vq_as1(g), vq_as3(s), 0,  0);
    __builtin_amdgcn_global_load_async_to_lds_b128(vq_as1(g), vq_as3(s), 16, 0);
    __builtin_amdgcn_global_load_async_to_lds_b128(vq_as1(g), vq_as3(s), 32, 0);
    __builtin_amdgcn_global_load_async_to_lds_b128(vq_as1(g), vq_as3(s), 48, 0);
#else
#pragma unroll
    for (int i = 0; i < 4; ++i)
      *(v8h*)(s + 8 * i) = *(const v8h*)(g + 8 * i);
#endif
  };

  // B fragment: row = codebook row n16 of this tile; halves = K{0..7} then
  // K{16..23} for lanes 0-15, +8 for lanes 16-31 (same order as A).
  auto loadB = [&](const _Float16* brow, int kt) -> v16h {
    v8h b0 = *(const v8h*)(brow + kt * 32);
    v8h b1 = *(const v8h*)(brow + kt * 32 + 16);
    return __builtin_shufflevector(b0, b1,
        0, 1, 2, 3, 4, 5, 6, 7, 8, 9, 10, 11, 12, 13, 14, 15);
  };

  const int ntiles = M >> 4;
  stage(0, 0);
  vq_wait_async();
  __syncthreads();

  for (int t = 0; t < ntiles; ++t) {
    const int buf = t & 1;
    if (t + 1 < ntiles) stage(t + 1, buf ^ 1);  // overlaps with WMMA burst

    const _Float16* brow = &lds[buf][n16 * LDS_STRIDE + (hi ? 8 : 0)];
    v8f acc = {};
#pragma unroll
    for (int kt = 0; kt < 16; ++kt) {
      v16h b = loadB(brow, kt);
      acc = __builtin_amdgcn_wmma_f32_16x16x32_f16(
          false, a[kt], false, b, (short)0, acc, false, false);
    }
    vq_sched_pipeline();  // DS reads 4 fragments ahead of WMMA consumption

    const int jcol = t * 16 + n16;
#pragma unroll
    for (int r = 0; r < 8; ++r)
      if (acc[r] > best[r]) { best[r] = acc[r]; bidx[r] = jcol; }

    vq_wait_async();   // next tile's DMA landed in LDS
    __syncthreads();
  }

  // C/D layout: VGPR r -> x-row (row_base + r) in lanes 0-15,
  // (row_base + 8 + r) in lanes 16-31; column = lane%16.
  // Reduce over the 16 lanes of each half (xor masks stay in-half).
#pragma unroll
  for (int r = 0; r < 8; ++r) {
    float b_ = best[r];
    int   i_ = bidx[r];
#pragma unroll
    for (int m = 8; m >= 1; m >>= 1) {
      float ob = __shfl_xor(b_, m, 32);
      int   oi = __shfl_xor(i_, m, 32);
      if (ob > b_ || (ob == b_ && oi < i_)) { b_ = ob; i_ = oi; }
    }
    if (lane == 0)       best_idx[row_base + r]     = i_;
    else if (lane == 16) best_idx[row_base + 8 + r] = i_;
  }
}

// ---------------------------------------------------------------------------
// Kernel 4: q_st[n] = cb_norm[idx[n]]; loss[n] = 1 - <flat[n], cb_norm[idx[n]]>
// recomputed in f32 for accuracy. One wave per row.
// ---------------------------------------------------------------------------
__global__ void __launch_bounds__(256) vq_gather(
    const float* __restrict__ x, const float* __restrict__ cbn,
    const int* __restrict__ best_idx, float* __restrict__ q_st,
    float* __restrict__ loss, int N) {
  const int wave = threadIdx.x >> 5;
  const int lane = threadIdx.x & 31;
  const int row  = blockIdx.x * 8 + wave;
  if (row >= N) return;

  const int j = best_idx[row];
  const float4* xs = (const float4*)(x   + (size_t)row * DIM);
  const float4* cs = (const float4*)(cbn + (size_t)j   * DIM);
  float4*       qd = (float4*)(q_st + (size_t)row * DIM);

  float ss = 0.f, dot = 0.f;
  float4 cr[4];
#pragma unroll
  for (int i = 0; i < 4; ++i) {
    float4 xv = xs[lane + 32 * i];
    cr[i] = cs[lane + 32 * i];
    ss  += xv.x * xv.x + xv.y * xv.y + xv.z * xv.z + xv.w * xv.w;
    dot += xv.x * cr[i].x + xv.y * cr[i].y + xv.z * cr[i].z + xv.w * cr[i].w;
  }
#pragma unroll
  for (int m = 16; m >= 1; m >>= 1) {
    ss  += __shfl_xor(ss, m, 32);
    dot += __shfl_xor(dot, m, 32);
  }
#pragma unroll
  for (int i = 0; i < 4; ++i) qd[lane + 32 * i] = cr[i];

  if (lane == 0) {
    const float scale = 1.0f / fmaxf(sqrtf(ss), 1e-12f);
    loss[row] = 1.0f - dot * scale;
  }
}

// ---------------------------------------------------------------------------
extern "C" void kernel_launch(void* const* d_in, const int* in_sizes, int n_in,
                              void* d_out, int out_size, void* d_ws, size_t ws_size,
                              hipStream_t stream) {
  (void)n_in; (void)out_size; (void)ws_size;
  const float* x  = (const float*)d_in[0];
  const float* cb = (const float*)d_in[1];
  const int N = in_sizes[0] / DIM;   // 32768
  const int M = in_sizes[1] / DIM;   // 4096

  // d_out = q_st [N*512] ++ commitment_loss [N] ++ cb_normalized [M*512]
  float* q_st   = (float*)d_out;
  float* loss   = q_st + (size_t)N * DIM;
  float* cb_out = loss + N;

  // workspace: flat_f16 [N*512] ++ cb_f16 [M*512] ++ best_idx [N]
  char* ws = (char*)d_ws;
  _Float16* flat16 = (_Float16*)ws;
  _Float16* cb16   = (_Float16*)(ws + (size_t)N * DIM * sizeof(_Float16));
  int* bidx        = (int*)(ws + ((size_t)N + (size_t)M) * DIM * sizeof(_Float16));

  vq_norm_rows<<<(M + 7) / 8, 256, 0, stream>>>(cb, cb_out, cb16, M);
  vq_norm_rows<<<(N + 7) / 8, 256, 0, stream>>>(x, nullptr, flat16, N);
  vq_gemm_argmax<<<N / 128, 256, 0, stream>>>(flat16, cb16, bidx, M);
  vq_gather<<<(N + 7) / 8, 256, 0, stream>>>(x, cb_out, bidx, q_st, loss, N);
}
